// Block_51797305589862
// MI455X (gfx1250) — compile-verified, matrix operand order checked by reading, not run on hardware
//
#include <hip/hip_runtime.h>
#include <hip/hip_bf16.h>
#include <math.h>

// ---- problem constants ----
#define BB   64
#define SS   577
#define DD   768
#define HH   12
#define DHD  64
#define FFF  3072
#define MR   (BB*SS)      // 36928 rows, divisible by 64
#define SPAD 608          // 19*32, padded seq length for V^T
#define SKT  19           // ceil(577/32) key tiles
#define QT   10           // ceil(577/64) query tiles

typedef __attribute__((ext_vector_type(16))) _Float16 v16h;
typedef __attribute__((ext_vector_type(8)))  _Float16 v8h;
typedef __attribute__((ext_vector_type(8)))  float    v8f;

__device__ __forceinline__ v16h cat8(v8h lo, v8h hi) {
  return __builtin_shufflevector(lo, hi, 0,1,2,3,4,5,6,7,8,9,10,11,12,13,14,15);
}
__device__ __forceinline__ v8f wmma16(v16h a, v16h b, v8f c) {
  return __builtin_amdgcn_wmma_f32_16x16x32_f16(false, a, false, b, (short)0, c, false, false);
}

// ---------------- LayerNorm (f32 in -> f16 out) ----------------
__global__ __launch_bounds__(256) void ln_f16_kernel(
    const float* __restrict__ x, const float* __restrict__ g,
    const float* __restrict__ b, _Float16* __restrict__ out) {
  const int row = blockIdx.x;
  const int t = threadIdx.x;
  const float* xr = x + (size_t)row * DD;
  float v0 = xr[t], v1 = xr[t + 256], v2 = xr[t + 512];
  __shared__ float rs[256], rq[256];
  rs[t] = v0 + v1 + v2;
  rq[t] = v0*v0 + v1*v1 + v2*v2;
  __syncthreads();
  for (int o = 128; o > 0; o >>= 1) {
    if (t < o) { rs[t] += rs[t + o]; rq[t] += rq[t + o]; }
    __syncthreads();
  }
  const float mu = rs[0] * (1.0f / DD);
  const float var = rq[0] * (1.0f / DD) - mu * mu;
  const float rstd = rsqrtf(var + 1e-6f);
  _Float16* o16 = out + (size_t)row * DD;
  o16[t]       = (_Float16)((v0 - mu) * rstd * g[t]       + b[t]);
  o16[t + 256] = (_Float16)((v1 - mu) * rstd * g[t + 256] + b[t + 256]);
  o16[t + 512] = (_Float16)((v2 - mu) * rstd * g[t + 512] + b[t + 512]);
}

// ---------------- weight conversion (all weights stored TRANSPOSED, f16) ----
// generic [K,N] f32 -> [N,K] f16
__global__ void cvt_t_kernel(const float* __restrict__ in, _Float16* __restrict__ out,
                             int K, int N) {
  int i = blockIdx.x * 256 + threadIdx.x;
  if (i >= K * N) return;
  int n = i % N, k = i / N;
  out[(size_t)n * K + k] = (_Float16)in[i];
}
// [H, D, DH] f32 -> WT [H*DH, D] f16  (head-concat N-dim becomes the row index)
__global__ void cvt_qkvw_t_kernel(const float* __restrict__ in, _Float16* __restrict__ out) {
  int i = blockIdx.x * 256 + threadIdx.x;
  if (i >= HH * DD * DHD) return;
  int e = i % DHD;
  int d = (i / DHD) % DD;
  int h = i / (DHD * DD);
  out[((size_t)(h * DHD + e)) * DD + d] = (_Float16)in[i];
}

// ---------------- WMMA GEMM, no LDS: fragments straight from L1/L2 ----------
// C[M,N] = A_f16[M,K] @ (WT[N,K])^T + bias, epilogue by EPI:
//  0: store f16   1: tanh-GELU, store f16   2: +residual f32, store f32
//  3: store f16 transposed as V^T[n*B + b][SPAD] for flash-attention
// block: 256 threads = 8 waves (2 waveM x 4 waveN), wave tile 32x64,
// block tile 64 x 256. K step 32 -> 12 global_load_b128 : 8 WMMA per wave.
template <int EPI, int KDIM, int NDIM>
__global__ __launch_bounds__(256) void gemm_kernel(
    const _Float16* __restrict__ A, const _Float16* __restrict__ WT,
    const float* __restrict__ bias, const float* __restrict__ res,
    void* __restrict__ out) {
  const int tid = threadIdx.x;
  const int lane = tid & 31, wave = tid >> 5;
  const int lmod = lane & 15, lhi = lane >> 4;
  const int waveM = wave & 1, waveN = wave >> 1;     // 2 x 4 wave grid
  const int m0 = blockIdx.y * 64 + waveM * 32;
  const int n0 = blockIdx.x * 256 + waveN * 64;
  const int kb = lhi << 3;

  const _Float16* a0 = A + (size_t)(m0 + lmod) * KDIM + kb;
  const _Float16* a1 = A + (size_t)(m0 + 16 + lmod) * KDIM + kb;
  const _Float16* w0 = WT + (size_t)(n0 + lmod) * KDIM + kb;

  v8f acc[2][4] = {};

  for (int k0 = 0; k0 < KDIM; k0 += 32) {
    v16h af0 = cat8(*(const v8h*)(a0 + k0), *(const v8h*)(a0 + k0 + 16));
    v16h af1 = cat8(*(const v8h*)(a1 + k0), *(const v8h*)(a1 + k0 + 16));
#pragma unroll
    for (int ni = 0; ni < 4; ni++) {
      const _Float16* wp = w0 + (size_t)ni * 16 * KDIM + k0;
      v16h bf = cat8(*(const v8h*)(wp), *(const v8h*)(wp + 16));
      acc[0][ni] = wmma16(af0, bf, acc[0][ni]);
      acc[1][ni] = wmma16(af1, bf, acc[1][ni]);
    }
  }

#pragma unroll
  for (int ni = 0; ni < 4; ni++) {
    const int n = n0 + ni * 16 + lmod;
    const float bv = bias[n];
#pragma unroll
    for (int mf = 0; mf < 2; mf++) {
#pragma unroll
      for (int r = 0; r < 8; r++) {
        const int m = m0 + mf * 16 + r + (lhi << 3);
        float val = acc[mf][ni][r] + bv;
        if constexpr (EPI == 0) {
          ((_Float16*)out)[(size_t)m * NDIM + n] = (_Float16)val;
        } else if constexpr (EPI == 1) {
          const float c = 0.7978845608028654f;
          float gl = 0.5f * val * (1.0f + tanhf(c * (val + 0.044715f * val * val * val)));
          ((_Float16*)out)[(size_t)m * NDIM + n] = (_Float16)gl;
        } else if constexpr (EPI == 2) {
          ((float*)out)[(size_t)m * NDIM + n] = val + res[(size_t)m * NDIM + n];
        } else {  // EPI == 3: V^T layout [n][b][SPAD]
          int bb = m / SS, ss = m - bb * SS;
          ((_Float16*)out)[((size_t)n * BB + bb) * SPAD + ss] = (_Float16)val;
        }
      }
    }
  }
}

// ---------------- fused flash attention ----------------
// grid = (QT, H, B), block = 128 (4 waves, 16 query rows per wave)
__global__ __launch_bounds__(128) void attn_kernel(
    const _Float16* __restrict__ Q, const _Float16* __restrict__ K,
    const _Float16* __restrict__ VT, _Float16* __restrict__ O) {
  __shared__ float Sb[4][16][33];
  __shared__ alignas(16) _Float16 Pb[4][16][40];
  __shared__ float alpha_s[4][16];
  __shared__ float l_s[4][16];

  const int tid = threadIdx.x;
  const int lane = tid & 31, w = tid >> 5;
  const int lmod = lane & 15, lhi = lane >> 4;
  const int kb = lhi << 3;
  const int qt = blockIdx.x, hh = blockIdx.y, bz = blockIdx.z;

  const int qrow_raw = qt * 64 + w * 16 + lmod;
  const int qrow = qrow_raw < SS ? qrow_raw : SS - 1;
  const size_t qoff = ((size_t)(bz * SS + qrow)) * DD + hh * DHD;

  v16h qf0 = cat8(*(const v8h*)(Q + qoff + kb),      *(const v8h*)(Q + qoff + kb + 16));
  v16h qf1 = cat8(*(const v8h*)(Q + qoff + 32 + kb), *(const v8h*)(Q + qoff + 32 + kb + 16));

  v8f o[4] = {};
  float mreg = -3.0e38f;   // row running max (valid in lanes 0..15)
  float lreg = 0.0f;       // row running sum

  for (int j = 0; j < SKT; j++) {
    // ---- scores S = Q K^T / 8, 16 x 32 tile ----
#pragma unroll
    for (int c = 0; c < 2; c++) {
      int t_raw = j * 32 + c * 16 + lmod;
      int t = t_raw < SS ? t_raw : SS - 1;
      const size_t koff = ((size_t)(bz * SS + t)) * DD + hh * DHD;
      v16h kf0 = cat8(*(const v8h*)(K + koff + kb),      *(const v8h*)(K + koff + kb + 16));
      v16h kf1 = cat8(*(const v8h*)(K + koff + 32 + kb), *(const v8h*)(K + koff + 32 + kb + 16));
      v8f sc = {};
      sc = wmma16(qf0, kf0, sc);
      sc = wmma16(qf1, kf1, sc);
#pragma unroll
      for (int r = 0; r < 8; r++) {
        int row = r + (lhi << 3);
        int col = c * 16 + lmod;
        float sv = sc[r] * 0.125f;
        if (j * 32 + col >= SS) sv = -3.0e38f;
        Sb[w][row][col] = sv;
      }
    }
    // ---- online softmax (per-wave LDS; in-order DS within a wave) ----
    if (lane < 16) {
      const int row = lane;
      float mj = mreg;
      for (int c2 = 0; c2 < 32; c2++) mj = fmaxf(mj, Sb[w][row][c2]);
      float al = __expf(mreg - mj);
      float sum = 0.0f;
      for (int c2 = 0; c2 < 32; c2++) {
        float p = __expf(Sb[w][row][c2] - mj);
        sum += p;
        Pb[w][row][c2] = (_Float16)p;
      }
      lreg = al * lreg + sum;
      mreg = mj;
      alpha_s[w][row] = al;
      l_s[w][row] = lreg;
    }
    // ---- rescale accumulated O by alpha ----
#pragma unroll
    for (int r = 0; r < 8; r++) {
      float al = alpha_s[w][r + (lhi << 3)];
#pragma unroll
      for (int c = 0; c < 4; c++) o[c][r] *= al;
    }
    // ---- O += P @ V (V^T layout gives contiguous B-fragments) ----
    const _Float16* pp = &Pb[w][lmod][kb];
    v16h pfrag = cat8(*(const v8h*)pp, *(const v8h*)(pp + 16));
#pragma unroll
    for (int c = 0; c < 4; c++) {
      int hd = c * 16 + lmod;
      const size_t voff = ((size_t)(hh * DHD + hd) * BB + bz) * SPAD + j * 32 + kb;
      v16h vf = cat8(*(const v8h*)(VT + voff), *(const v8h*)(VT + voff + 16));
      o[c] = wmma16(pfrag, vf, o[c]);
    }
  }

  // ---- normalize and store (head-concat f16 layout) ----
  const int qr0 = qt * 64 + w * 16;
#pragma unroll
  for (int r = 0; r < 8; r++) {
    const int row = r + (lhi << 3);
    const int q = qr0 + row;
    if (q < SS) {
      float inv = 1.0f / l_s[w][row];
#pragma unroll
      for (int c = 0; c < 4; c++) {
        O[((size_t)(bz * SS + q)) * DD + hh * DHD + c * 16 + lmod] = (_Float16)(o[c][r] * inv);
      }
    }
  }
}

// ---------------- host launcher ----------------
extern "C" void kernel_launch(void* const* d_in, const int* in_sizes, int n_in,
                              void* d_out, int out_size, void* d_ws, size_t ws_size,
                              hipStream_t stream) {
  (void)in_sizes; (void)n_in; (void)out_size; (void)ws_size;
  const float* x     = (const float*)d_in[0];
  const float* ln1_g = (const float*)d_in[1];
  const float* ln1_b = (const float*)d_in[2];
  const float* Wq    = (const float*)d_in[3];
  const float* bq    = (const float*)d_in[4];
  const float* Wk    = (const float*)d_in[5];
  const float* bk    = (const float*)d_in[6];
  const float* Wv    = (const float*)d_in[7];
  const float* bv    = (const float*)d_in[8];
  const float* Wo    = (const float*)d_in[9];
  const float* bo    = (const float*)d_in[10];
  const float* ln2_g = (const float*)d_in[11];
  const float* ln2_b = (const float*)d_in[12];
  const float* W1    = (const float*)d_in[13];
  const float* b1    = (const float*)d_in[14];
  const float* W2    = (const float*)d_in[15];
  const float* b2    = (const float*)d_in[16];

  char* ws = (char*)d_ws;
  size_t off = 0;
  auto take = [&](size_t bytes) -> char* {
    char* p = ws + off;
    off += (bytes + 255) & ~(size_t)255;
    return p;
  };
  _Float16* h16    = (_Float16*)take((size_t)MR * DD * 2);          // LN1 / LN2 output
  _Float16* q16    = (_Float16*)take((size_t)MR * DD * 2);
  _Float16* k16    = (_Float16*)take((size_t)MR * DD * 2);
  _Float16* vT16   = (_Float16*)take((size_t)DD * BB * SPAD * 2);
  _Float16* attn16 = (_Float16*)take((size_t)MR * DD * 2);
  float*    x1     = (float*)take((size_t)MR * DD * 4);             // post-attention residual
  _Float16* wqT    = (_Float16*)take((size_t)DD * DD * 2);
  _Float16* wkT    = (_Float16*)take((size_t)DD * DD * 2);
  _Float16* wvT    = (_Float16*)take((size_t)DD * DD * 2);
  _Float16* woT    = (_Float16*)take((size_t)DD * DD * 2);
  _Float16* w1T    = (_Float16*)take((size_t)DD * FFF * 2);
  _Float16* w2T    = (_Float16*)take((size_t)FFF * DD * 2);
  // FF intermediate (226.9 MB) aliases dead q/k/vT/attn span (229.9 MB)
  _Float16* g16 = q16;
  _Float16* m16 = h16;

  // weight conversion (to transposed f16)
  {
    int n = HH * DD * DHD;
    cvt_qkvw_t_kernel<<<(n + 255) / 256, 256, 0, stream>>>(Wq, wqT);
    cvt_qkvw_t_kernel<<<(n + 255) / 256, 256, 0, stream>>>(Wk, wkT);
    cvt_qkvw_t_kernel<<<(n + 255) / 256, 256, 0, stream>>>(Wv, wvT);
    n = DD * DD;
    cvt_t_kernel<<<(n + 255) / 256, 256, 0, stream>>>(Wo, woT, DD, DD);
    n = DD * FFF;
    cvt_t_kernel<<<(n + 255) / 256, 256, 0, stream>>>(W1, w1T, DD, FFF);
    cvt_t_kernel<<<(n + 255) / 256, 256, 0, stream>>>(W2, w2T, FFF, DD);
  }

  // zero V^T so the padded tail can never inject NaN/Inf into P@V
  hipMemsetAsync(vT16, 0, (size_t)DD * BB * SPAD * 2, stream);

  // LN1
  ln_f16_kernel<<<MR, 256, 0, stream>>>(x, ln1_g, ln1_b, h16);

  // QKV projections (block tile 64 x 256)
  dim3 gD(DD / 256, MR / 64);       // (3, 577)
  gemm_kernel<0, DD, DD><<<gD, 256, 0, stream>>>(h16, wqT, bq, nullptr, q16);
  gemm_kernel<0, DD, DD><<<gD, 256, 0, stream>>>(h16, wkT, bk, nullptr, k16);
  gemm_kernel<3, DD, DD><<<gD, 256, 0, stream>>>(h16, wvT, bv, nullptr, vT16);

  // fused attention
  dim3 gA(QT, HH, BB);
  attn_kernel<<<gA, 128, 0, stream>>>(q16, k16, vT16, attn16);

  // output projection + residual
  gemm_kernel<2, DD, DD><<<gD, 256, 0, stream>>>(attn16, woT, bo, x, x1);

  // LN2
  ln_f16_kernel<<<MR, 256, 0, stream>>>(x1, ln2_g, ln2_b, m16);

  // MLP
  dim3 gF(FFF / 256, MR / 64);      // (12, 577)
  gemm_kernel<1, DD, FFF><<<gF, 256, 0, stream>>>(m16, w1T, b1, nullptr, g16);
  gemm_kernel<2, FFF, DD><<<gD, 256, 0, stream>>>(g16, w2T, b2, x1, (float*)d_out);
}
